// QuantizedLinearINT6_26731876451112
// MI455X (gfx1250) — compile-verified, matrix operand order checked by reading, not run on hardware
//
#include <hip/hip_runtime.h>
#include <hip/hip_bf16.h>

// ---------------------------------------------------------------------------
// QuantizedLinearINT6: y[b,o] = scale[o] * sum_k x[b,k]*w_int6[o,k] + bias[o]
// M=8(pad 16), N=12288 (768 tiles of 16), K=4096 (split-K 4 x 1024 per block).
// HBM-bound on the 201 MB int32 weight stream; WMMA f16->f32 does the MACs.
// ---------------------------------------------------------------------------

#define IN_FEATURES   4096
#define OUT_FEATURES  12288
#define BATCH         8
#define TILE_N        16
#define N_TILES       (OUT_FEATURES / TILE_N)   // 768
#define SPLIT_K       4
#define K_PER_WAVE    (IN_FEATURES / SPLIT_K)   // 1024
#define K_STEP        32
#define K_ITERS       (K_PER_WAVE / K_STEP)     // 32

typedef __attribute__((ext_vector_type(16))) _Float16 v16h;
typedef __attribute__((ext_vector_type(8)))  _Float16 v8h;
typedef __attribute__((ext_vector_type(8)))  float    v8f;
typedef __attribute__((ext_vector_type(8)))  int      v8i;

// Convert x (8 x 4096 fp32) into a zero-padded 16 x 4096 f16 matrix in ws.
// Rows 8..15 are zero because the WMMA A-fragment reads all 16 rows.
__global__ __launch_bounds__(256) void qlin_prep_x(const float* __restrict__ x,
                                                   _Float16* __restrict__ xh) {
    int idx = blockIdx.x * 256 + threadIdx.x;        // 0 .. 16*4096-1
    int row = idx >> 12;
    int col = idx & (IN_FEATURES - 1);
    float v = (row < BATCH) ? x[(size_t)row * IN_FEATURES + col] : 0.0f;
    xh[idx] = (_Float16)v;
}

__global__ __launch_bounds__(128) void qlin_main(const _Float16* __restrict__ xh,
                                                 const int*      __restrict__ w,
                                                 const float*    __restrict__ scale,
                                                 const float*    __restrict__ bias,
                                                 float*          __restrict__ out) {
    __shared__ float red[SPLIT_K * BATCH * TILE_N];  // 4 * 8 * 16 floats

    const int ntile    = blockIdx.x;                 // one 16-wide N tile per block
    const int wave     = threadIdx.x >> 5;           // split-K part 0..3
    const int lane     = threadIdx.x & 31;
    const int half_sel = lane >> 4;                  // lane group 0 / 1
    const int nl       = lane & 15;                  // row/col within 16-tile

    const int oc               = ntile * TILE_N + nl;          // output channel
    const _Float16* __restrict__ xrow = xh + (size_t)nl * IN_FEATURES; // A row m=lane&15
    const int*      __restrict__ wrow = w  + (size_t)oc * IN_FEATURES; // B column = channel
    const int kbase = wave * K_PER_WAVE;

    v8f c = {};

#pragma unroll 4
    for (int kk = 0; kk < K_ITERS; ++kk) {
        const int k = kbase + kk * K_STEP;
        // ISA 16-bit A 16x32 layout: lane holds K = [h*8 .. h*8+7] and
        // [16+h*8 .. 16+h*8+7]  (h = lane>>4).  B (32x16 col-major) identical
        // pattern with the output channel as the 16-wide dimension.
        const int koff0 = k + half_sel * 8;
        const int koff1 = k + 16 + half_sel * 8;

        v8h a0 = *(const v8h*)(xrow + koff0);        // global_load_b128
        v8h a1 = *(const v8h*)(xrow + koff1);
        v8i w0 = *(const v8i*)(wrow + koff0);        // 2x global_load_b128 each
        v8i w1 = *(const v8i*)(wrow + koff1);

        // keep the weight stream ahead of the machine (~1KB lookahead)
        __builtin_prefetch((const void*)(wrow + k + 256), 0, 1);

        v16h a, b;
#pragma unroll
        for (int i = 0; i < 8; ++i) {
            a[i]     = a0[i];
            a[i + 8] = a1[i];
            b[i]     = (_Float16)(short)w0[i];       // int6 exact in f16
            b[i + 8] = (_Float16)(short)w1[i];
        }

        c = __builtin_amdgcn_wmma_f32_16x16x32_f16(
                /*neg_a=*/false, a, /*neg_b=*/false, b,
                /*c_mod=*/(short)0, c, /*reuse_a=*/false, /*reuse_b=*/false);
    }

    // C-fragment: VGPR r -> row r (lanes 0-15) / row r+8 (lanes 16-31).
    // Only rows 0..7 are real batch rows -> only lanes 0-15 contribute.
    if (half_sel == 0) {
#pragma unroll
        for (int r = 0; r < BATCH; ++r)
            red[wave * (BATCH * TILE_N) + r * TILE_N + nl] = c[r];
    }
    __syncthreads();

    // Deterministic split-K reduction: thread t -> (row, col) of the 8x16 tile.
    const int r = threadIdx.x >> 4;                  // 0..7
    const int n = threadIdx.x & 15;                  // 0..15
    float sum = red[0 * (BATCH * TILE_N) + r * TILE_N + n]
              + red[1 * (BATCH * TILE_N) + r * TILE_N + n]
              + red[2 * (BATCH * TILE_N) + r * TILE_N + n]
              + red[3 * (BATCH * TILE_N) + r * TILE_N + n];
    const int o = ntile * TILE_N + n;
    out[(size_t)r * OUT_FEATURES + o] = sum * scale[o] + bias[o];
}

extern "C" void kernel_launch(void* const* d_in, const int* in_sizes, int n_in,
                              void* d_out, int out_size, void* d_ws, size_t ws_size,
                              hipStream_t stream) {
    const float* x     = (const float*)d_in[0];  // [8, 4096] fp32
    const int*   w     = (const int*)  d_in[1];  // [12288, 4096] int32 (int6 values)
    const float* scale = (const float*)d_in[2];  // [12288] fp32
    const float* bias  = (const float*)d_in[3];  // [12288] fp32
    float* out = (float*)d_out;                  // [8, 12288] fp32

    _Float16* xh = (_Float16*)d_ws;              // 16 x 4096 f16 = 128 KB scratch

    qlin_prep_x<<<(16 * IN_FEATURES) / 256, 256, 0, stream>>>(x, xh);
    qlin_main<<<N_TILES, 128, 0, stream>>>(xh, w, scale, bias, out);
}
